// MultiHeadAttention_44246753083948
// MI455X (gfx1250) — matrix operand order repaired and checked by reading, not compiled, not run on hardware
//
#include <hip/hip_runtime.h>
#include <hip/hip_bf16.h>

// ---------------- problem constants (match reference) ----------------
constexpr int BATCH  = 4;
constexpr int SEQ    = 2048;
constexpr int DMODEL = 1024;
constexpr int NH     = 16;
constexpr int DK     = 64;          // DMODEL / NH
constexpr int MTOK   = BATCH * SEQ; // 8192 rows total

typedef __attribute__((ext_vector_type(16))) _Float16 v16h;
typedef __attribute__((ext_vector_type(8)))  _Float16 v8h;
typedef __attribute__((ext_vector_type(8)))  float    v8f;

union F16x16 { v16h v; v8h h8[2]; };

// ---------------- WMMA fragment loaders (CDNA5 16-bit layouts) ----------------
// A 16x32 f16: lane L: row = L&15; halves[0..7]=K(ksel*8+0..7),
// halves[8..15]=K(16+ksel*8+0..7) where ksel = L>>4. Two contiguous 16B loads.
__device__ __forceinline__ v16h load_a_frag(const _Float16* A, int ld) {
  const int lane = threadIdx.x & 31;
  const int row  = lane & 15;
  const int ksel = lane >> 4;
  const _Float16* p = A + (size_t)row * ld + ksel * 8;
  F16x16 u;
  u.h8[0] = *(const v8h*)(p);
  u.h8[1] = *(const v8h*)(p + 16);
  return u.v;
}

// B 32x16 f16 where source is stored [n][k] (k contiguous): lane L: col = L&15,
// halves[0..7] = src[col][kg*8 + 0..7], halves[8..15] = src[col][16 + kg*8 + 0..7].
__device__ __forceinline__ v16h load_b_frag(const _Float16* Bs, int ld) {
  const int lane = threadIdx.x & 31;
  const int col  = lane & 15;
  const int kg   = lane >> 4;
  const _Float16* p = Bs + (size_t)col * ld + kg * 8;
  F16x16 u;
  u.h8[0] = *(const v8h*)(p);
  u.h8[1] = *(const v8h*)(p + 16);
  return u.v;
}

__device__ __forceinline__ v8f wmma_f16(v16h a, v16h b, v8f c) {
  return __builtin_amdgcn_wmma_f32_16x16x32_f16(
      /*neg_a=*/false, a, /*neg_b=*/false, b,
      /*c_mod=*/(short)0, c, /*reuse_a=*/false, /*reuse_b=*/false);
}

// ---------------- f32 -> f16 convert ----------------
__global__ void cvt_f32_f16(const float* __restrict__ in, _Float16* __restrict__ out, int n) {
  int i = blockIdx.x * blockDim.x + threadIdx.x;
  const int stride = gridDim.x * blockDim.x;
  for (; i < n; i += stride) out[i] = (_Float16)in[i];
}

// ---------------- GEMM epilogue for one 16x16 tile ----------------
// MODE 1: f16 out, head-split [B,H,SEQ,DK]     (Q, K projections)
// MODE 2: f16 out, transposed [B,H,DK,SEQ]     (V projection)
// MODE 3: f32 out, plain      [M, DMODEL]      (output projection)
template <int MODE>
__device__ __forceinline__ void store_tile(const v8f& acc, int m0, int n0,
                                           const float* __restrict__ bias,
                                           void* __restrict__ out) {
  const int lane = threadIdx.x & 31;
  const int col  = n0 + (lane & 15);
  const float bv = bias[col];
#pragma unroll
  for (int i = 0; i < 8; ++i) {
    const int row = m0 + i + 8 * (lane >> 4);
    const float v = acc[i] + bv;
    if constexpr (MODE == 1 || MODE == 2) {
      const int b = row >> 11, n = row & (SEQ - 1);
      const int h = col >> 6,  dk = col & (DK - 1);
      _Float16* o = (_Float16*)out;
      const size_t idx = (MODE == 1)
          ? ((((size_t)(b * NH + h)) * SEQ + n) * DK + dk)   // [B,H,N,DK]
          : ((((size_t)(b * NH + h)) * DK + dk) * SEQ + n);  // [B,H,DK,N]
      o[idx] = (_Float16)v;
    } else {
      ((float*)out)[(size_t)row * DMODEL + col] = v;
    }
  }
}

// ---------------- WMMA GEMM: C[M,N] = A[M,K] @ W[N,K]^T + bias ----------------
// 256 threads = 8 waves arranged 2(M) x 4(N); each wave register-blocks 2x2
// WMMA tiles (32x32 output) -> 4 WMMAs per 4 fragment loads (2x the FLOP/byte
// of a 1x1 tile). Block tile = 64 x 128.
template <int MODE>
__global__ void gemm16(const _Float16* __restrict__ A, const _Float16* __restrict__ Bw,
                       const float* __restrict__ bias, void* __restrict__ out, int K) {
  const int wave = threadIdx.x >> 5;
  const int m0 = blockIdx.x * 64  + (wave >> 2) * 32;
  const int n0 = blockIdx.y * 128 + (wave & 3) * 32;

  const _Float16* a0p = A  + (size_t)m0 * K;
  const _Float16* a1p = a0p + (size_t)16 * K;
  const _Float16* b0p = Bw + (size_t)n0 * K;
  const _Float16* b1p = b0p + (size_t)16 * K;

  v8f acc00 = {}, acc01 = {}, acc10 = {}, acc11 = {};
#pragma unroll 2
  for (int k = 0; k < K; k += 32) {
    const v16h af0 = load_a_frag(a0p + k, K);
    const v16h af1 = load_a_frag(a1p + k, K);
    const v16h bf0 = load_b_frag(b0p + k, K);
    const v16h bf1 = load_b_frag(b1p + k, K);
    acc00 = wmma_f16(af0, bf0, acc00);
    acc01 = wmma_f16(af0, bf1, acc01);
    acc10 = wmma_f16(af1, bf0, acc10);
    acc11 = wmma_f16(af1, bf1, acc11);
  }

  store_tile<MODE>(acc00, m0,      n0,      bias, out);
  store_tile<MODE>(acc01, m0,      n0 + 16, bias, out);
  store_tile<MODE>(acc10, m0 + 16, n0,      bias, out);
  store_tile<MODE>(acc11, m0 + 16, n0 + 16, bias, out);
}

// ---------------- fused attention: scores -> softmax -> P@V ----------------
// One block = one (b,h, 16-row q strip). 256 threads = 8 waves.
// Dynamic LDS: scores f32 [16][2048] | probs f16 [16][2048] | red[256]+row[16] | pv f32 [16][64]
__global__ void attn_kernel(const _Float16* __restrict__ q, const _Float16* __restrict__ k,
                            const _Float16* __restrict__ vt, float* __restrict__ probs_out,
                            _Float16* __restrict__ ctx) {
  const int tid  = threadIdx.x;
  const int wave = tid >> 5, lane = tid & 31;
  const int qt = blockIdx.x & (SEQ / 16 - 1);  // 0..127
  const int bh = blockIdx.x >> 7;              // b*NH + h, 0..63
  const int bb = bh >> 4, hh = bh & 15;
  const int row0 = qt * 16;

  extern __shared__ char smem[];
  float*    sS   = (float*)smem;                                     // 16*2048 f32
  _Float16* sP   = (_Float16*)(smem + 16 * SEQ * 4);                 // 16*2048 f16
  float*    sRed = (float*)(smem + 16 * SEQ * 4 + 16 * SEQ * 2);     // 256
  float*    sRow = sRed + 256;                                       // 16
  float*    sAcc = sRow + 16;                                        // 16*64 f32

  const _Float16* qb = q  + (size_t)bh * SEQ * DK + (size_t)row0 * DK;
  const _Float16* kb = k  + (size_t)bh * SEQ * DK;
  const _Float16* vb = vt + (size_t)bh * DK * SEQ;

  // ---- phase 1: S = Q Kt / sqrt(dk), strip of 16 rows into LDS ----
  const v16h a0 = load_a_frag(qb +  0, DK);
  const v16h a1 = load_a_frag(qb + 32, DK);
  for (int t = wave; t < SEQ / 16; t += 8) {
    const int n0 = t * 16;
    const v16h b0 = load_b_frag(kb + (size_t)n0 * DK +  0, DK);
    const v16h b1 = load_b_frag(kb + (size_t)n0 * DK + 32, DK);
    v8f acc = {};
    acc = wmma_f16(a0, b0, acc);
    acc = wmma_f16(a1, b1, acc);
#pragma unroll
    for (int i = 0; i < 8; ++i)
      sS[(size_t)(i + 8 * (lane >> 4)) * SEQ + n0 + (lane & 15)] = acc[i] * 0.125f;
  }
  __syncthreads();

  // ---- phase 2: softmax, 16 threads per row ----
  const int rid = tid >> 4, sub = tid & 15;
  const float* srow = sS + (size_t)rid * SEQ;

  float m = -1e30f;
  for (int c = sub; c < SEQ; c += 16) m = fmaxf(m, srow[c]);
  sRed[tid] = m;
  __syncthreads();
  if (sub == 0) {
    float mm = sRed[rid * 16];
    for (int j = 1; j < 16; ++j) mm = fmaxf(mm, sRed[rid * 16 + j]);
    sRow[rid] = mm;
  }
  __syncthreads();
  const float rmax = sRow[rid];

  float s = 0.f;
  for (int c = sub; c < SEQ; c += 16) s += __expf(srow[c] - rmax);
  sRed[tid] = s;
  __syncthreads();
  if (sub == 0) {
    float ss = 0.f;
    for (int j = 0; j < 16; ++j) ss += sRed[rid * 16 + j];
    sRow[rid] = ss;
  }
  __syncthreads();
  const float rinv = 1.f / sRow[rid];

  float* prow = probs_out + ((size_t)bh * SEQ + row0 + rid) * SEQ;  // single write of attn
  _Float16* pl = sP + (size_t)rid * SEQ;
  for (int c = sub; c < SEQ; c += 16) {
    const float p = __expf(srow[c] - rmax) * rinv;
    prow[c] = p;
    pl[c]   = (_Float16)p;
  }
  for (int i2 = tid; i2 < 16 * DK; i2 += 256) sAcc[i2] = 0.f;
  __syncthreads();

  // ---- phase 3: O = P @ V  (K=2048 split across 8 waves, LDS reduce) ----
  const int kbase = wave * (SEQ / 8);  // 256 per wave
  v8f acc4[4] = {{}, {}, {}, {}};
  for (int kk = 0; kk < SEQ / 8; kk += 32) {
    const v16h pa = load_a_frag(sP + kbase + kk, SEQ);
#pragma unroll
    for (int nt = 0; nt < 4; ++nt) {
      const v16h vf = load_b_frag(vb + (size_t)(nt * 16) * SEQ + kbase + kk, SEQ);
      acc4[nt] = wmma_f16(pa, vf, acc4[nt]);
    }
  }
#pragma unroll
  for (int nt = 0; nt < 4; ++nt)
#pragma unroll
    for (int i = 0; i < 8; ++i)
      atomicAdd(&sAcc[(size_t)(i + 8 * (lane >> 4)) * DK + nt * 16 + (lane & 15)], acc4[nt][i]);
  __syncthreads();

  // ---- phase 4: ctx[b, n, h*64+dk] (f16, feeds output projection) ----
  for (int i2 = tid; i2 < 16 * DK; i2 += 256) {
    const int r = i2 >> 6, dk = i2 & 63;
    ctx[((size_t)bb * SEQ + row0 + r) * DMODEL + hh * DK + dk] = (_Float16)sAcc[i2];
  }
}

// ---------------- residual + LayerNorm ----------------
__global__ void resid_ln(const float* __restrict__ x, const float* __restrict__ proj,
                         const float* __restrict__ gamma, const float* __restrict__ beta,
                         float* __restrict__ out) {
  const int row = blockIdx.x;
  const float* xr = x    + (size_t)row * DMODEL;
  const float* pr = proj + (size_t)row * DMODEL;
  float* orow     = out  + (size_t)row * DMODEL;
  __shared__ float r1[256];
  __shared__ float r2[256];
  float s = 0.f, s2 = 0.f;
  for (int c = threadIdx.x; c < DMODEL; c += 256) {
    const float h = xr[c] + pr[c];
    s += h; s2 += h * h;
  }
  r1[threadIdx.x] = s; r2[threadIdx.x] = s2;
  __syncthreads();
  for (int off = 128; off > 0; off >>= 1) {
    if (threadIdx.x < off) {
      r1[threadIdx.x] += r1[threadIdx.x + off];
      r2[threadIdx.x] += r2[threadIdx.x + off];
    }
    __syncthreads();
  }
  const float mu   = r1[0] * (1.f / DMODEL);
  const float var  = r2[0] * (1.f / DMODEL) - mu * mu;
  const float rstd = rsqrtf(var + 1e-5f);
  for (int c = threadIdx.x; c < DMODEL; c += 256) {
    const float h = xr[c] + pr[c];
    orow[c] = (h - mu) * rstd * gamma[c] + beta[c];
  }
}

// ---------------- launch ----------------
extern "C" void kernel_launch(void* const* d_in, const int* in_sizes, int n_in,
                              void* d_out, int out_size, void* d_ws, size_t ws_size,
                              hipStream_t stream) {
  (void)in_sizes; (void)n_in; (void)out_size; (void)ws_size;
  const float* x     = (const float*)d_in[0];
  const float* Wq    = (const float*)d_in[1];
  const float* bq    = (const float*)d_in[2];
  const float* Wk    = (const float*)d_in[3];
  const float* bk    = (const float*)d_in[4];
  const float* Wv    = (const float*)d_in[5];
  const float* bv    = (const float*)d_in[6];
  const float* Wo    = (const float*)d_in[7];
  const float* bo    = (const float*)d_in[8];
  const float* gamma = (const float*)d_in[9];
  const float* beta  = (const float*)d_in[10];

  // workspace layout (bytes)
  char* ws = (char*)d_ws;
  size_t off = 0;
  _Float16* xh   = (_Float16*)(ws + off); off += (size_t)MTOK * DMODEL * 2;     // 16 MB
  _Float16* wqh  = (_Float16*)(ws + off); off += (size_t)DMODEL * DMODEL * 2;   // 2 MB
  _Float16* wkh  = (_Float16*)(ws + off); off += (size_t)DMODEL * DMODEL * 2;
  _Float16* wvh  = (_Float16*)(ws + off); off += (size_t)DMODEL * DMODEL * 2;
  _Float16* woh  = (_Float16*)(ws + off); off += (size_t)DMODEL * DMODEL * 2;
  _Float16* qh   = (_Float16*)(ws + off); off += (size_t)MTOK * DMODEL * 2;     // [B,H,N,DK]
  _Float16* kh   = (_Float16*)(ws + off); off += (size_t)MTOK * DMODEL * 2;     // [B,H,N,DK]
  _Float16* vth  = (_Float16*)(ws + off); off += (size_t)MTOK * DMODEL * 2;     // [B,H,DK,N]
  _Float16* ctxh = (_Float16*)(ws + off); off += (size_t)MTOK * DMODEL * 2;     // [M,D]
  float*    projf= (float*)(ws + off);    off += (size_t)MTOK * DMODEL * 4;     // 32 MB

  const size_t LNSZ = (size_t)MTOK * DMODEL;            // out_ln elements
  float* probs = (float*)d_out + LNSZ;                  // attn tensor region
  float* outln = (float*)d_out;

  // 1) convert inputs to f16
  cvt_f32_f16<<<512, 256, 0, stream>>>(x,  xh,  MTOK * DMODEL);
  cvt_f32_f16<<<256, 256, 0, stream>>>(Wq, wqh, DMODEL * DMODEL);
  cvt_f32_f16<<<256, 256, 0, stream>>>(Wk, wkh, DMODEL * DMODEL);
  cvt_f32_f16<<<256, 256, 0, stream>>>(Wv, wvh, DMODEL * DMODEL);
  cvt_f32_f16<<<256, 256, 0, stream>>>(Wo, woh, DMODEL * DMODEL);

  // 2) Q/K/V projections (WMMA): Q,K head-split; V transposed for P@V B-fragments
  const dim3 ggrid(MTOK / 64, DMODEL / 128);
  gemm16<1><<<ggrid, 256, 0, stream>>>(xh, wqh, bq, qh, DMODEL);
  gemm16<1><<<ggrid, 256, 0, stream>>>(xh, wkh, bk, kh, DMODEL);
  gemm16<2><<<ggrid, 256, 0, stream>>>(xh, wvh, bv, vth, DMODEL);

  // 3) fused attention (scores -> softmax -> P@V), attn written exactly once
  const int smemBytes = 16 * SEQ * 4 + 16 * SEQ * 2 + (256 + 16) * 4 + 16 * DK * 4;
  (void)hipFuncSetAttribute((const void*)attn_kernel,
                            hipFuncAttributeMaxDynamicSharedMemorySize, smemBytes);
  attn_kernel<<<BATCH * NH * (SEQ / 16), 256, smemBytes, stream>>>(qh, kh, vth, probs, ctxh);

  // 4) output projection (WMMA, f32 out)
  gemm16<3><<<ggrid, 256, 0, stream>>>(ctxh, woh, bo, projf, DMODEL);

  // 5) residual + LayerNorm
  resid_ln<<<MTOK, 256, 0, stream>>>(x, projf, gamma, beta, outln);
}
